// GINVirtual_node_9242769621977
// MI455X (gfx1250) — matrix operand order repaired and checked by reading, not compiled, not run on hardware
//
#include <hip/hip_runtime.h>
#include <hip/hip_bf16.h>

// ---------------------------------------------------------------------------
// GIN + virtual node forward for MI455X (gfx1250, wave32, WMMA).
// Input order (setup_inputs dict insertion order, params nested in order):
//  0 x[50000,9] i32        1 edge_index[2,600000] i32   2 edge_attr[600000,3] i32
//  3 batch[50000] i32      4 atom_emb[9,64,128] f32     5 bond_emb[5,3,8,128] f32
//  6 eps[5] f32            7 conv_w1[5,128,256]         8 conv_b1[5,256]
//  9 conv_bn1_g[5,256]    10 conv_bn1_b[5,256]         11 conv_w2[5,256,128]
// 12 conv_b2[5,128]       13 bn_g[5,128]               14 bn_b[5,128]
// 15 vn_emb[1,128]        16 vn_w1[4,128,256]          17 vn_b1[4,256]
// 18 vn_bn1_g[4,256]      19 vn_bn1_b[4,256]           20 vn_w2[4,256,128]
// 21 vn_b2[4,128]         22 vn_bn2_g[4,128]           23 vn_bn2_b[4,128]
// Output: h [50000,128] f32  (kept in d_out throughout)
// ---------------------------------------------------------------------------

#define N_NODES  50000
#define N_EDGES  600000
#define N_GRAPHS 256
#define EMB      128
#define NLAYERS  5

typedef __attribute__((ext_vector_type(4)))  __bf16 v4bf;
typedef __attribute__((ext_vector_type(8)))  __bf16 v8bf;
typedef __attribute__((ext_vector_type(16))) __bf16 v16bf;
typedef __attribute__((ext_vector_type(8)))  float  v8f;

// ----------------------------- elementwise kernels -------------------------

__global__ __launch_bounds__(256) void atom_encode_kernel(
    const int* __restrict__ x, const float* __restrict__ emb,
    float* __restrict__ h)
{
    int tid = blockIdx.x * 256 + threadIdx.x;          // N*128 threads
    int n = tid >> 7, c = tid & 127;
    float s = 0.f;
#pragma unroll
    for (int f = 0; f < 9; ++f) {
        int idx = x[n * 9 + f];
        s += emb[(f * 64 + idx) * EMB + c];
    }
    h[tid] = s;
}

__global__ __launch_bounds__(256) void vn_init_kernel(
    float* __restrict__ vn, const float* __restrict__ vn_emb)
{
    int tid = blockIdx.x * 256 + threadIdx.x;          // G*128 threads
    vn[tid] = vn_emb[tid & 127];
}

__global__ __launch_bounds__(256) void zero4_kernel(float4* __restrict__ p)
{
    int tid = blockIdx.x * 256 + threadIdx.x;
    p[tid] = make_float4(0.f, 0.f, 0.f, 0.f);
}

__global__ __launch_bounds__(256) void vn_add_kernel(
    float* __restrict__ h, const float* __restrict__ vn,
    const int* __restrict__ batch)
{
    int tid = blockIdx.x * 256 + threadIdx.x;          // N*32 threads
    int n = tid >> 5, c = (tid & 31) << 2;
    float4 v = *(const float4*)(vn + (batch[n] << 7) + c);
    float4* hp = (float4*)(h + (n << 7) + c);
    float4 hv = *hp;
    hv.x += v.x; hv.y += v.y; hv.z += v.z; hv.w += v.w;
    *hp = hv;
}

// relu(h[src] + bond_emb) scatter-added to agg[dst]; L2-resident atomics.
__global__ __launch_bounds__(256) void edge_scatter_kernel(
    const int* __restrict__ ei, const int* __restrict__ ea,
    const float* __restrict__ bond, const float* __restrict__ h,
    float* __restrict__ agg)
{
    int tid = blockIdx.x * 256 + threadIdx.x;          // E*32 threads
    int e = tid >> 5, c = (tid & 31) << 2;
    int src = ei[e], dst = ei[N_EDGES + e];
    float4 m = *(const float4*)(h + (src << 7) + c);
#pragma unroll
    for (int f = 0; f < 3; ++f) {
        int idx = ea[e * 3 + f];
        float4 bv = *(const float4*)(bond + ((f * 8 + idx) << 7) + c);
        m.x += bv.x; m.y += bv.y; m.z += bv.z; m.w += bv.w;
    }
    m.x = fmaxf(m.x, 0.f); m.y = fmaxf(m.y, 0.f);
    m.z = fmaxf(m.z, 0.f); m.w = fmaxf(m.w, 0.f);
    float* ap = agg + (dst << 7) + c;
    atomicAdd(ap + 0, m.x); atomicAdd(ap + 1, m.y);
    atomicAdd(ap + 2, m.z); atomicAdd(ap + 3, m.w);
}

// global_add_pool: pooled[batch[n]] += h[n]
__global__ __launch_bounds__(256) void pool_scatter_kernel(
    const float* __restrict__ h, const int* __restrict__ batch,
    float* __restrict__ pooled)
{
    int tid = blockIdx.x * 256 + threadIdx.x;          // N*32 threads
    int n = tid >> 5, c = (tid & 31) << 2;
    float4 v = *(const float4*)(h + (n << 7) + c);
    float* pp = pooled + (batch[n] << 7) + c;
    atomicAdd(pp + 0, v.x); atomicAdd(pp + 1, v.y);
    atomicAdd(pp + 2, v.z); atomicAdd(pp + 3, v.w);
}

// zb = bf16((1+eps)*h + agg)
__global__ __launch_bounds__(256) void z_convert_kernel(
    const float* __restrict__ h, const float* __restrict__ agg,
    const float* __restrict__ eps, int l, __bf16* __restrict__ zb)
{
    int tid = blockIdx.x * 256 + threadIdx.x;          // N*128
    float e = 1.0f + eps[l];
    zb[tid] = (__bf16)(fmaf(e, h[tid], agg[tid]));
}

// pooledb = bf16(pooled + vn)
__global__ __launch_bounds__(256) void pool_convert_kernel(
    const float* __restrict__ pooled, const float* __restrict__ vn,
    __bf16* __restrict__ pooledb)
{
    int tid = blockIdx.x * 256 + threadIdx.x;          // G*128
    pooledb[tid] = (__bf16)(pooled[tid] + vn[tid]);
}

// --------------------------- WMMA GEMM + BN epilogue ------------------------
// Out[M,NOUT] = epilogue(A[M,K](bf16) @ W[K,NOUT](f32) + bias) with eval BN.
// Block: 128 threads = 4 waves. Block tile: 16 rows x 128 cols; wave w owns
// cols [w*32, w*32+32) as TWO 16x16 WMMA tiles sharing the A fragment.
// Grid: (M/16, NOUT/128). K staged through LDS in 64-wide chunks.

template <int K, int NOUT, bool RELU, bool OUT_BF16>
__global__ __launch_bounds__(128) void gemm_bn_kernel(
    const __bf16* __restrict__ A, const float* __restrict__ W,
    const float* __restrict__ bias, const float* __restrict__ gamma,
    const float* __restrict__ beta, void* __restrict__ Out)
{
    constexpr int KB = 64;
    __shared__ __bf16 As[16 * KB];       // 2 KB
    __shared__ __bf16 Bs[KB * 128];      // 16 KB

    const int t    = threadIdx.x;        // 0..127
    const int lane = t & 31;
    const int w    = t >> 5;             // wave -> 32-col subtile
    const int half = lane >> 4;
    const int ml   = lane & 15;
    const int rowBase = blockIdx.x * 16;
    const int colBase = blockIdx.y * 128;

    // loop-invariant staging addresses (strength-reduced: per-chunk/iteration
    // deltas are compile-time constants -> folded into instruction offsets)
    const __bf16* aSrc = A + (size_t)(rowBase + (t >> 3)) * K + ((t & 7) << 3);
    __bf16*       aDst = &As[(t >> 3) * KB + ((t & 7) << 3)];
    const float*  wSrc = W + (size_t)(t >> 5) * NOUT + colBase + ((t & 31) << 2);
    __bf16*       bDst = &Bs[(t >> 5) * 128 + ((t & 31) << 2)];

    v8f acc0 = {};
    v8f acc1 = {};

    for (int k0 = 0; k0 < K; k0 += KB) {
        // ---- stage A tile 16 x 64: one 16B vector per thread ----
        *(v8bf*)aDst = *(const v8bf*)(aSrc + k0);
        // ---- stage B tile 64 x 128: float4 loads, pack to bf16 ----
        {
            const float* wp = wSrc + (size_t)k0 * NOUT;
#pragma unroll
            for (int i = 0; i < 16; ++i) {             // rows (t>>5) + 4*i
                float4 f = *(const float4*)(wp + i * 4 * NOUT);
                v4bf bb;
                bb[0] = (__bf16)f.x; bb[1] = (__bf16)f.y;
                bb[2] = (__bf16)f.z; bb[3] = (__bf16)f.w;
                *(v4bf*)(bDst + i * 4 * 128) = bb;
            }
        }
        if (k0 + KB < K) {
            // prefetch next A chunk (global_prefetch_b8)
            __builtin_prefetch(aSrc + k0 + KB, 0, 1);
        }
        __syncthreads();

#pragma unroll
        for (int ks = 0; ks < KB; ks += 32) {
            // A fragment (16-bit 16x32): lane m=ml, K = {8h+0..7, 16+8h+0..7}
            const __bf16* ap = &As[ml * KB + ks + 8 * half];
            v8bf a0 = *(const v8bf*)ap;
            v8bf a1 = *(const v8bf*)(ap + 16);
            // B fragments (16-bit 32x16): lane n=ml, K = {8h+0..7, 16+8h+0..7}
            const __bf16* bp = &Bs[(ks + 8 * half) * 128 + w * 32 + ml];
            v16bf av, bv0, bv1;
#pragma unroll
            for (int i = 0; i < 8; ++i) {
                av[i]  = a0[i];  av[i + 8]  = a1[i];
                bv0[i] = bp[i * 128]; bv0[i + 8] = bp[(i + 16) * 128];
                bv1[i] = bp[i * 128 + 16]; bv1[i + 8] = bp[(i + 16) * 128 + 16];
            }
            acc0 = __builtin_amdgcn_wmma_f32_16x16x32_bf16(
                false, av, false, bv0, (short)0, acc0, false, false);
            acc1 = __builtin_amdgcn_wmma_f32_16x16x32_bf16(
                false, av, false, bv1, (short)0, acc1, false, false);
        }
        __syncthreads();
    }

    // ---- epilogue: bias + eval BatchNorm (mean=0, var=1) + optional relu ----
    const float inv = 0.999995000037499e0f;  // 1/sqrt(1+1e-5)
    int col0 = colBase + w * 32 + ml;
    int col1 = col0 + 16;
    float sc0 = gamma[col0] * inv, bi0 = bias[col0], be0 = beta[col0];
    float sc1 = gamma[col1] * inv, bi1 = bias[col1], be1 = beta[col1];
#pragma unroll
    for (int r = 0; r < 8; ++r) {
        int row = rowBase + r + 8 * half;   // C/D layout: VGPR r -> M=r(+8)
        float v0 = (acc0[r] + bi0) * sc0 + be0;
        float v1 = (acc1[r] + bi1) * sc1 + be1;
        if (RELU) {
            v0 = v0 > 0.f ? v0 : 0.f;
            v1 = v1 > 0.f ? v1 : 0.f;
        }
        if (OUT_BF16) {
            ((__bf16*)Out)[(size_t)row * NOUT + col0] = (__bf16)v0;
            ((__bf16*)Out)[(size_t)row * NOUT + col1] = (__bf16)v1;
        } else {
            ((float*)Out)[(size_t)row * NOUT + col0] = v0;
            ((float*)Out)[(size_t)row * NOUT + col1] = v1;
        }
    }
}

// ------------------------------- host driver --------------------------------

extern "C" void kernel_launch(void* const* d_in, const int* in_sizes, int n_in,
                              void* d_out, int out_size, void* d_ws, size_t ws_size,
                              hipStream_t stream)
{
    const int*   x         = (const int*)d_in[0];
    const int*   ei        = (const int*)d_in[1];
    const int*   ea        = (const int*)d_in[2];
    const int*   batch     = (const int*)d_in[3];
    const float* atom_emb  = (const float*)d_in[4];
    const float* bond_emb  = (const float*)d_in[5];
    const float* eps       = (const float*)d_in[6];
    const float* conv_w1   = (const float*)d_in[7];
    const float* conv_b1   = (const float*)d_in[8];
    const float* bn1_g     = (const float*)d_in[9];
    const float* bn1_b     = (const float*)d_in[10];
    const float* conv_w2   = (const float*)d_in[11];
    const float* conv_b2   = (const float*)d_in[12];
    const float* bn_g      = (const float*)d_in[13];
    const float* bn_b      = (const float*)d_in[14];
    const float* vn_emb    = (const float*)d_in[15];
    const float* vn_w1     = (const float*)d_in[16];
    const float* vn_b1     = (const float*)d_in[17];
    const float* vn_bn1_g  = (const float*)d_in[18];
    const float* vn_bn1_b  = (const float*)d_in[19];
    const float* vn_w2     = (const float*)d_in[20];
    const float* vn_b2     = (const float*)d_in[21];
    const float* vn_bn2_g  = (const float*)d_in[22];
    const float* vn_bn2_b  = (const float*)d_in[23];

    float* h = (float*)d_out;                          // [N, 128] lives in d_out

    // carve workspace
    char* p = (char*)d_ws;
    auto carve = [&](size_t bytes) {
        void* r = (void*)p;
        p += (bytes + 255) & ~(size_t)255;
        return r;
    };
    float*  agg     = (float*)carve((size_t)N_NODES * EMB * 4);     // 25.6 MB
    __bf16* zb      = (__bf16*)carve((size_t)N_NODES * EMB * 2);    // 12.8 MB
    __bf16* yb      = (__bf16*)carve((size_t)N_NODES * 256 * 2);    // 25.6 MB
    float*  pooled  = (float*)carve((size_t)N_GRAPHS * EMB * 4);
    __bf16* pooledb = (__bf16*)carve((size_t)N_GRAPHS * EMB * 2);
    __bf16* vtmpb   = (__bf16*)carve((size_t)N_GRAPHS * 256 * 2);
    float*  vnbuf   = (float*)carve((size_t)N_GRAPHS * EMB * 4);
    (void)in_sizes; (void)n_in; (void)out_size; (void)ws_size;

    const int gNE   = (N_NODES * EMB) / 256;           // 25000
    const int gN32  = (N_NODES * 32) / 256;            // 6250
    const int gE32  = (N_EDGES * 32) / 256;            // 75000
    const int gGE   = (N_GRAPHS * EMB) / 256;          // 128

    atom_encode_kernel<<<gNE, 256, 0, stream>>>(x, atom_emb, h);
    vn_init_kernel<<<gGE, 256, 0, stream>>>(vnbuf, vn_emb);

    for (int l = 0; l < NLAYERS; ++l) {
        // h += vn[batch]
        vn_add_kernel<<<gN32, 256, 0, stream>>>(h, vnbuf, batch);

        // agg = scatter_add(relu(h[src] + bond_emb), dst)
        zero4_kernel<<<(N_NODES * EMB / 4) / 256, 256, 0, stream>>>((float4*)agg);
        edge_scatter_kernel<<<gE32, 256, 0, stream>>>(
            ei, ea, bond_emb + (size_t)l * 3 * 8 * EMB, h, agg);

        // pooled = segment_sum(h, batch)  (before h is overwritten)
        if (l < NLAYERS - 1) {
            zero4_kernel<<<(N_GRAPHS * EMB / 4) / 256, 256, 0, stream>>>((float4*)pooled);
            pool_scatter_kernel<<<gN32, 256, 0, stream>>>(h, batch, pooled);
        }

        // z = (1+eps)*h + agg  -> bf16
        z_convert_kernel<<<gNE, 256, 0, stream>>>(h, agg, eps, l, zb);

        // y = relu(BN(z @ W1 + b1))  [N,256] bf16
        gemm_bn_kernel<128, 256, true, true>
            <<<dim3(N_NODES / 16, 2), 128, 0, stream>>>(
            zb, conv_w1 + (size_t)l * 128 * 256, conv_b1 + l * 256,
            bn1_g + l * 256, bn1_b + l * 256, yb);

        // hn = BN(y @ W2 + b2) (+relu unless last layer) -> h (f32, d_out)
        if (l < NLAYERS - 1)
            gemm_bn_kernel<256, 128, true, false>
                <<<dim3(N_NODES / 16, 1), 128, 0, stream>>>(
                yb, conv_w2 + (size_t)l * 256 * 128, conv_b2 + l * 128,
                bn_g + l * 128, bn_b + l * 128, h);
        else
            gemm_bn_kernel<256, 128, false, false>
                <<<dim3(N_NODES / 16, 1), 128, 0, stream>>>(
                yb, conv_w2 + (size_t)l * 256 * 128, conv_b2 + l * 128,
                bn_g + l * 128, bn_b + l * 128, h);

        // virtual-node MLP update
        if (l < NLAYERS - 1) {
            pool_convert_kernel<<<gGE, 256, 0, stream>>>(pooled, vnbuf, pooledb);
            gemm_bn_kernel<128, 256, true, true>
                <<<dim3(N_GRAPHS / 16, 2), 128, 0, stream>>>(
                pooledb, vn_w1 + (size_t)l * 128 * 256, vn_b1 + l * 256,
                vn_bn1_g + l * 256, vn_bn1_b + l * 256, vtmpb);
            gemm_bn_kernel<256, 128, true, false>
                <<<dim3(N_GRAPHS / 16, 1), 128, 0, stream>>>(
                vtmpb, vn_w2 + (size_t)l * 256 * 128, vn_b2 + l * 128,
                vn_bn2_g + l * 128, vn_bn2_b + l * 128, vnbuf);
        }
    }
}
